// MGIN_42021960024258
// MI455X (gfx1250) — matrix operand order repaired and checked by reading, not compiled, not run on hardware
//
#include <hip/hip_runtime.h>
#include <hip/hip_bf16.h>

// ---------- types for WMMA / TDM ----------
typedef __attribute__((ext_vector_type(16))) __bf16 v16bf;
typedef __attribute__((ext_vector_type(8)))  __bf16 v8bf;
typedef __attribute__((ext_vector_type(8)))  float  v8f;
typedef __attribute__((ext_vector_type(4)))  unsigned int u32x4;
typedef __attribute__((ext_vector_type(8)))  int    i32x8;
typedef __attribute__((ext_vector_type(4)))  int    i32x4;

union AV {
    v16bf v;
    v8bf  h[2];
};

#define N_NODES 10000
#define N_EDGES 160000
#define DIM     1280
#define ND      (N_NODES * DIM)          // 12,800,000

// ---------- workspace layout (bytes) ----------
#define WS_AGG   ((size_t)0)                         // f32  [N,D]   51,200,000
#define WS_XB    ((size_t)51200000)                  // bf16 [N,D]   25,600,000
#define WS_H     ((size_t)76800000)                  // f32  [N,D]   51,200,000
#define WS_WB1   ((size_t)128000000)                 // bf16 [D,D]    3,276,800
#define WS_WB2   ((size_t)131276800)                 // bf16 [D,D]    3,276,800
#define WS_EW    ((size_t)134553600)                 // f32  [E]        640,000

// ---------- hardware f32 atomic add (no CAS loop) ----------
__device__ __forceinline__ void gatomic_add_f32(float* p, float v) {
    asm volatile("global_atomic_add_f32 %0, %1, off" :: "v"(p), "v"(v) : "memory");
}

// ---------- TDM: load a 64-row x 128-col bf16 tile (row stride 1280) into LDS ----------
// Descriptor per ISA 8.3/8.4: group0 {count=1, lds_addr, global_addr, type=2},
// group1 {data_size=2B, tensor_dim0=128, tensor_dim1=64, tile_dim0=128, tile_dim1=64,
//         tensor_dim0_stride=1280}. Groups 2/3 zero (2D tensor).
// This toolchain's builtin takes 6 args: (u32x4, i32x8, i32x4, i32x4, i32x8, i32 cpol).
__device__ __forceinline__ void tdm_load_w_chunk(unsigned lds_addr, const __bf16* gptr) {
    unsigned long long ga = (unsigned long long)(size_t)gptr;
    u32x4 g0;
    g0[0] = 1u;                                              // count=1 (valid user D#)
    g0[1] = lds_addr;                                        // LDS byte address
    g0[2] = (unsigned)(ga & 0xFFFFFFFFull);                  // global_addr[31:0]
    g0[3] = (unsigned)((ga >> 32) & 0x01FFFFFFull)           // global_addr[56:32]
            | 0x80000000u;                                   // type=2 ("image")
    i32x8 g1;
    g1[0] = 0x00010000;            // data_size=1 -> 2 bytes/elem
    g1[1] = (int)(128u << 16);     // tensor_dim0 = 128 (bits 63:48)
    g1[2] = (int)(64u  << 16);     // tensor_dim1 = 64  (bits 95:80)
    g1[3] = (int)(128u << 16);     // tile_dim0   = 128 (bits 127:112)
    g1[4] = 64;                    // tile_dim1   = 64  (bits 143:128)
    g1[5] = 1280;                  // tensor_dim0_stride = 1280 elems (bits 191:160)
    g1[6] = 0;
    g1[7] = 0;
    i32x4 z4 = {0, 0, 0, 0};
    i32x8 z8 = {0, 0, 0, 0, 0, 0, 0, 0};
    __builtin_amdgcn_tensor_load_to_lds(g0, g1, z4, z4, z8, 0);
}

// ---------------- edge weights: 1/(w^2 + 1e-6) ----------------
__global__ void ew_kernel(const float* __restrict__ ef, float* __restrict__ ew, int E) {
    int e = blockIdx.x * blockDim.x + threadIdx.x;
    if (e < E) {
        float w = ef[e];
        ew[e] = 1.0f / (w * w + 1e-6f);
    }
}

// ---------------- f32 -> bf16 weight conversion ----------------
__global__ void wcvt_kernel(const float* __restrict__ w, __bf16* __restrict__ wb, int n) {
    int i = blockIdx.x * blockDim.x + threadIdx.x;
    if (i < n) wb[i] = (__bf16)w[i];
}

// ---------------- zero f32 buffer (float4 granularity) ----------------
__global__ void zero4_kernel(float4* __restrict__ p, int n4) {
    int i = blockIdx.x * blockDim.x + threadIdx.x;
    if (i < n4) p[i] = make_float4(0.f, 0.f, 0.f, 0.f);
}

__global__ void zero1_kernel(float* __restrict__ p, int n) {
    int i = blockIdx.x * blockDim.x + threadIdx.x;
    if (i < n) p[i] = 0.f;
}

// ---------------- edge scatter: agg[dst] += ew[e] * feat[src] ----------------
// one block (320 threads = 10 waves) per edge; each thread owns 4 floats
__global__ void scatter_kernel(const float* __restrict__ feat, const int* __restrict__ ei,
                               const float* __restrict__ ew, float* __restrict__ agg) {
    const int e = blockIdx.x;
    const int t = threadIdx.x;            // 0..319
    const int src = ei[e];
    const int dst = ei[N_EDGES + e];
    const float w = ew[e];
    const float4 v = *(const float4*)(feat + (size_t)src * DIM + t * 4);
    float* ap = agg + (size_t)dst * DIM + t * 4;
    gatomic_add_f32(ap + 0, v.x * w);
    gatomic_add_f32(ap + 1, v.y * w);
    gatomic_add_f32(ap + 2, v.z * w);
    gatomic_add_f32(ap + 3, v.w * w);
}

// ---------------- x_bf16 = bf16(feat + agg) ----------------
__global__ void cvt_add_kernel(const float* __restrict__ feat, const float* __restrict__ agg,
                               __bf16* __restrict__ xb, int n4) {
    int i = blockIdx.x * blockDim.x + threadIdx.x;
    if (i >= n4) return;
    const float4 a = ((const float4*)feat)[i];
    const float4 b = ((const float4*)agg)[i];
    __bf16* o = xb + (size_t)i * 4;
    o[0] = (__bf16)(a.x + b.x);
    o[1] = (__bf16)(a.y + b.y);
    o[2] = (__bf16)(a.z + b.z);
    o[3] = (__bf16)(a.w + b.w);
}

// ---------------- WMMA GEMM with TDM-staged B panel ----------------
// out[M,Dn] = X[M,Dk](bf16) @ W[Dn,Dk]^T(bf16) + bias.
// Block = 128 threads (4 waves) = one 16-row M-tile x 64-col N-panel.
// Wave w owns N-subtile [ntg*64 + w*16 .. +15].
// W panel (64 x 1280) is streamed into LDS as ten 64x128 chunks by the
// Tensor Data Mover, double-buffered (TENSORcnt + workgroup barriers).
// A lanes: lane l holds row m=l&15, k = (l>>4)*16..+15 (contiguous).
// B lanes: lane l holds col n=l&15, k = (l>>4)*16..+15 (contiguous, from LDS).
// C/D: VGPR r -> M = r + 8*(l>=16), N = l&15.
__global__ void gemm_bf16_wmma_tdm_kernel(const __bf16* __restrict__ X,
                                          const __bf16* __restrict__ W,
                                          const float* __restrict__ bias,
                                          float* __restrict__ out) {
    __shared__ __align__(64) __bf16 lds[2][64][128];   // 2 x 16 KB double buffer

    const int lane = threadIdx.x & 31;
    const int wave = threadIdx.x >> 5;        // 0..3
    const int mt   = blockIdx.x / 20;         // 0..624  (625*16 == 10000 exactly)
    const int ntg  = blockIdx.x % 20;         // 0..19   (20*64  == 1280 exactly)
    const int lmod = lane & 15;
    const int kgrp = lane >> 4;               // 0 or 1

    const __bf16* xrow  = X + (size_t)(mt * 16 + lmod) * DIM + kgrp * 16;
    const __bf16* wbase = W + (size_t)(ntg * 64) * DIM;        // panel base (k=0)
    const unsigned lds_base = (unsigned)(size_t)(void*)&lds[0][0][0];

    if (wave == 0) {                                   // wave-uniform; TDM ignores EXEC
        tdm_load_w_chunk(lds_base, wbase);             // chunk 0 -> buf 0
    }

    v8f c = {};
    for (int ch = 0; ch < 10; ++ch) {
        if (wave == 0) {
            if (ch + 1 < 10) {
                tdm_load_w_chunk(lds_base + (unsigned)(((ch + 1) & 1) * 16384),
                                 wbase + (ch + 1) * 128);
                __builtin_amdgcn_s_wait_tensorcnt(1);  // chunk ch landed in LDS
            } else {
                __builtin_amdgcn_s_wait_tensorcnt(0);  // last chunk landed
            }
        }
        __syncthreads();                               // LDS chunk visible to all waves

        const __bf16* bp = &lds[ch & 1][wave * 16 + lmod][kgrp * 16];
        const __bf16* ap = xrow + ch * 128;
        __builtin_prefetch(ap + 128, 0, 1);            // next A chunk -> global_prefetch_b8
#pragma unroll
        for (int kk = 0; kk < 4; ++kk) {
            AV a, b;
            a.h[0] = *(const v8bf*)(ap + kk * 32);
            a.h[1] = *(const v8bf*)(ap + kk * 32 + 8);
            b.v    = *(const v16bf*)(bp + kk * 32);
            c = __builtin_amdgcn_wmma_f32_16x16x32_bf16(
                    /*neg_a=*/false, a.v, /*neg_b=*/false, b.v,
                    /*c_mod=*/(short)0, c, /*reuse_a=*/false, /*reuse_b=*/false);
        }
        __syncthreads();                               // done reading before buf reuse
    }

    const int col  = ntg * 64 + wave * 16 + lmod;
    const int mrow = mt * 16 + kgrp * 8;
    const float bv = bias[col];
#pragma unroll
    for (int r = 0; r < 8; ++r) {
        out[(size_t)(mrow + r) * DIM + col] = c[r] + bv;
    }
}

// ---------------- partial column sums: out[d] += sum_{n in chunk} (h[n,d] + f0[n,d]) ----------------
__global__ void mean_res_partial_kernel(const float* __restrict__ h, const float* __restrict__ f0,
                                        float* __restrict__ out) {
    const int d  = blockIdx.x * blockDim.x + threadIdx.x;   // 0..1279 (5 blocks x 256)
    const int n0 = blockIdx.y * 250;                        // 40 node chunks
    if (d >= DIM) return;
    float s = 0.f;
    for (int n = n0; n < n0 + 250; ++n) {
        s += h[(size_t)n * DIM + d] + f0[(size_t)n * DIM + d];
    }
    gatomic_add_f32(out + d, s * (1.0f / (float)N_NODES));
}

extern "C" void kernel_launch(void* const* d_in, const int* in_sizes, int n_in,
                              void* d_out, int out_size, void* d_ws, size_t ws_size,
                              hipStream_t stream) {
    const float* node_feat0 = (const float*)d_in[0];   // [N, D]
    const float* edge_feat  = (const float*)d_in[1];   // [E]
    const int*   edge_index = (const int*)d_in[2];     // [2, E]
    const float* W1         = (const float*)d_in[3];   // [D, D]
    const float* b1         = (const float*)d_in[4];   // [D]
    const float* W2         = (const float*)d_in[5];   // [D, D]
    const float* b2         = (const float*)d_in[6];   // [D]
    float* out = (float*)d_out;                        // [1, D]

    char* ws = (char*)d_ws;
    float*  agg = (float*)(ws + WS_AGG);
    __bf16* xb  = (__bf16*)(ws + WS_XB);
    float*  h   = (float*)(ws + WS_H);
    __bf16* wb1 = (__bf16*)(ws + WS_WB1);
    __bf16* wb2 = (__bf16*)(ws + WS_WB2);
    float*  ew  = (float*)(ws + WS_EW);

    // edge weights + bf16 weights (recomputed every launch: deterministic)
    ew_kernel<<<(N_EDGES + 255) / 256, 256, 0, stream>>>(edge_feat, ew, N_EDGES);
    wcvt_kernel<<<(DIM * DIM + 255) / 256, 256, 0, stream>>>(W1, wb1, DIM * DIM);
    wcvt_kernel<<<(DIM * DIM + 255) / 256, 256, 0, stream>>>(W2, wb2, DIM * DIM);

    const int n4 = ND / 4;              // 3,200,000
    const int gemm_blocks = 625 * 20;   // 12,500

    // ---- layer 1 ----
    zero4_kernel<<<(n4 + 255) / 256, 256, 0, stream>>>((float4*)agg, n4);
    scatter_kernel<<<N_EDGES, 320, 0, stream>>>(node_feat0, edge_index, ew, agg);
    cvt_add_kernel<<<(n4 + 255) / 256, 256, 0, stream>>>(node_feat0, agg, xb, n4);
    gemm_bf16_wmma_tdm_kernel<<<gemm_blocks, 128, 0, stream>>>(xb, wb1, b1, h);

    // ---- layer 2 ----
    zero4_kernel<<<(n4 + 255) / 256, 256, 0, stream>>>((float4*)agg, n4);
    scatter_kernel<<<N_EDGES, 320, 0, stream>>>(h, edge_index, ew, agg);
    cvt_add_kernel<<<(n4 + 255) / 256, 256, 0, stream>>>(h, agg, xb, n4);
    // h no longer needed as input after cvt; reuse as layer-2 output buffer
    gemm_bf16_wmma_tdm_kernel<<<gemm_blocks, 128, 0, stream>>>(xb, wb2, b2, h);

    // ---- residual + mean over nodes (two-stage, parallel over node chunks) ----
    zero1_kernel<<<(DIM + 255) / 256, 256, 0, stream>>>(out, DIM);
    dim3 mr_grid((DIM + 255) / 256, 40);
    mean_res_partial_kernel<<<mr_grid, 256, 0, stream>>>(h, node_feat0, out);
}
